// SpecDetrHead_15307263443665
// MI455X (gfx1250) — matrix operand order repaired and checked by reading, not compiled, not run on hardware
//
#include <hip/hip_runtime.h>
#include <hip/hip_bf16.h>

// ---------------------------------------------------------------------------
// SpecDETR head, fused, f16 WMMA (f32 accumulate) for MI455X / gfx1250.
//   L=6, B=32, Q=900, D=256, C=8.  M = B*Q = 28800 rows per layer.
// ---------------------------------------------------------------------------

#define NLAYERS 6
#define BQ      28800      // B*Q
#define DIM     256
#define NCLS    8
#define M_TILE  64         // rows per block
#define BLK_THREADS 128    // 4 waves (wave32)

typedef __attribute__((ext_vector_type(16))) _Float16 v16h;
typedef __attribute__((ext_vector_type(8)))  _Float16 v8h;
typedef __attribute__((ext_vector_type(4)))  _Float16 v4h;
typedef __attribute__((ext_vector_type(8)))  float    v8f;

// ---------------------------------------------------------------------------
// Weight repack: f32 [L, K=256, Nsrc] -> f16 WMMA B-fragments.
// Fragment = 32x16 halfs, stored [lane][h] contiguously (32 bytes/lane).
// B layout (V_WMMA_F32_16X16X32_F16, 32x16 KxN):
//   lanes 0..15 : col n = lane,    halfs 0..7 -> K = k0 + h, halfs 8..15 -> K = k0 + 16 + (h-8)
//   lanes 16..31: col n = lane-16, same with +8 K sub-blocks
// Fragment order per layer: frag_idx = n_tile * k_tiles + k_tile.
// ---------------------------------------------------------------------------
__global__ void pack_weights_f16(const float* __restrict__ W,
                                 _Float16* __restrict__ out,
                                 int Nsrc, int n_tiles, int k_tiles) {
    int per_layer = n_tiles * k_tiles;
    int frag  = blockIdx.x;
    int layer = frag / per_layer;
    int rem   = frag - layer * per_layer;
    int n_tile = rem / k_tiles;
    int k_tile = rem - n_tile * k_tiles;

    int lane = threadIdx.x >> 4;        // 0..31
    int h    = threadIdx.x & 15;        // 0..15
    int n    = n_tile * 16 + (lane & 15);
    int K    = k_tile * 32 + ((lane >> 4) << 3) + (h & 7) + ((h >> 3) << 4);

    float v = 0.0f;
    if (n < Nsrc)
        v = W[(size_t)layer * DIM * Nsrc + (size_t)K * Nsrc + n];

    out[((size_t)layer * per_layer + (size_t)(n_tile * k_tiles + k_tile)) * 512
        + lane * 16 + h] = (_Float16)v;
}

// ---------------------------------------------------------------------------
// A-fragment from row-major f16 LDS tile (row stride DIM).
// A layout (16x32 f16): lanes 0..15 row m=lane:  halfs 0..7 = K k0..k0+7,
//   halfs 8..15 = K k0+16..k0+23;  lanes 16..31 get the +8 K sub-blocks.
// ---------------------------------------------------------------------------
__device__ __forceinline__ v16h load_a_frag(const _Float16* lds, int m0, int k0, int lane) {
    int row  = m0 + (lane & 15);
    int base = row * DIM + k0 + ((lane >> 4) << 3);
    v8h lo = *(const v8h*)(lds + base);
    v8h hi = *(const v8h*)(lds + base + 16);
    return __builtin_shufflevector(lo, hi, 0,1,2,3,4,5,6,7,8,9,10,11,12,13,14,15);
}

__device__ __forceinline__ v16h load_b_frag(const _Float16* __restrict__ pack,
                                            int frag_idx, int lane) {
    const _Float16* p = pack + (size_t)frag_idx * 512 + lane * 16;
    v8h lo = *(const v8h*)(p);
    v8h hi = *(const v8h*)(p + 8);
    return __builtin_shufflevector(lo, hi, 0,1,2,3,4,5,6,7,8,9,10,11,12,13,14,15);
}

__device__ __forceinline__ v8f wmma_f16(v16h a, v16h b, v8f c) {
    return __builtin_amdgcn_wmma_f32_16x16x32_f16(false, a, false, b,
                                                  (short)0, c, false, false);
}

// ---------------------------------------------------------------------------
// Fused head kernel. Grid = NLAYERS * (BQ / M_TILE) blocks, 128 threads.
// Each wave (32 lanes) owns a 16-row strip of the 64-row tile.
// ---------------------------------------------------------------------------
__global__ __launch_bounds__(BLK_THREADS)
void spec_detr_head_kernel(const float* __restrict__ h_in,
                           const float* __restrict__ refs,
                           const _Float16* __restrict__ w1p,
                           const _Float16* __restrict__ w2p,
                           const _Float16* __restrict__ wcp,
                           const _Float16* __restrict__ w3p,
                           const float* __restrict__ b_cls,
                           const float* __restrict__ b1,
                           const float* __restrict__ b2,
                           const float* __restrict__ b3,
                           float* __restrict__ out_cls,
                           float* __restrict__ out_bbox) {
    __shared__ _Float16 sA[M_TILE * DIM];   // h (f16); later reused as t2
    __shared__ _Float16 sT[M_TILE * DIM];   // t1

    const int tiles_per_layer = BQ / M_TILE;           // 450
    const int layer = blockIdx.x / tiles_per_layer;
    const int tile  = blockIdx.x - layer * tiles_per_layer;
    const int row0  = tile * M_TILE;

    // ---- stage h tile: f32 global -> f16 LDS (flat 64x256 copy) ----
    const float* hsrc = h_in + (size_t)layer * BQ * DIM + (size_t)row0 * DIM;
    for (int i = threadIdx.x; i < (M_TILE * DIM) / 4; i += BLK_THREADS) {
        float4 f = ((const float4*)hsrc)[i];
        v4h q;
        q[0] = (_Float16)f.x; q[1] = (_Float16)f.y;
        q[2] = (_Float16)f.z; q[3] = (_Float16)f.w;
        *(v4h*)(sA + i * 4) = q;
    }
    __syncthreads();

    const int wave = threadIdx.x >> 5;
    const int lane = threadIdx.x & 31;
    const int m0   = wave * 16;                 // this wave's row strip in tile
    const int rhi  = (lane >> 4) << 3;          // +8 rows for upper half-wave
    const int ncol = lane & 15;

    // preload A fragments for the strip (h)
    v16h afr[8];
#pragma unroll
    for (int kt = 0; kt < 8; ++kt)
        afr[kt] = load_a_frag(sA, m0, kt * 32, lane);

    // ---- classification branch: h @ W_cls (N=8 padded to 16) ----
    {
        const _Float16* wL = wcp + (size_t)layer * 8 * 512;   // 1 n_tile, 8 k_tiles
        v8f acc = {};
#pragma unroll
        for (int kt = 0; kt < 8; ++kt)
            acc = wmma_f16(afr[kt], load_b_frag(wL, kt, lane), acc);
        if (ncol < NCLS) {
            float bias = b_cls[layer * NCLS + ncol];
#pragma unroll
            for (int v = 0; v < 8; ++v) {
                int gr = row0 + m0 + v + rhi;
                out_cls[((size_t)layer * BQ + gr) * NCLS + ncol] = acc[v] + bias;
            }
        }
    }

    // ---- FFN1: t1 = relu(h @ W1 + b1) -> sT ----
    {
        const _Float16* wL = w1p + (size_t)layer * 16 * 8 * 512;
#pragma unroll 1
        for (int nt = 0; nt < 16; ++nt) {
            v8f acc = {};
#pragma unroll
            for (int kt = 0; kt < 8; ++kt)
                acc = wmma_f16(afr[kt], load_b_frag(wL, nt * 8 + kt, lane), acc);
            int col = nt * 16 + ncol;
            float bias = b1[layer * DIM + col];
#pragma unroll
            for (int v = 0; v < 8; ++v) {
                int row = m0 + v + rhi;
                float x = acc[v] + bias;
                sT[row * DIM + col] = (_Float16)(x > 0.f ? x : 0.f);
            }
        }
    }

    // reload A fragments from t1 (wave-private rows: no barrier needed)
#pragma unroll
    for (int kt = 0; kt < 8; ++kt)
        afr[kt] = load_a_frag(sT, m0, kt * 32, lane);

    // ---- FFN2: t2 = relu(t1 @ W2 + b2) -> sA (reuse) ----
    {
        const _Float16* wL = w2p + (size_t)layer * 16 * 8 * 512;
#pragma unroll 1
        for (int nt = 0; nt < 16; ++nt) {
            v8f acc = {};
#pragma unroll
            for (int kt = 0; kt < 8; ++kt)
                acc = wmma_f16(afr[kt], load_b_frag(wL, nt * 8 + kt, lane), acc);
            int col = nt * 16 + ncol;
            float bias = b2[layer * DIM + col];
#pragma unroll
            for (int v = 0; v < 8; ++v) {
                int row = m0 + v + rhi;
                float x = acc[v] + bias;
                sA[row * DIM + col] = (_Float16)(x > 0.f ? x : 0.f);
            }
        }
    }

#pragma unroll
    for (int kt = 0; kt < 8; ++kt)
        afr[kt] = load_a_frag(sA, m0, kt * 32, lane);

    // ---- FFN3 + box refinement: sigmoid(reg + inv_sigmoid(ref)) ----
    {
        const _Float16* wL = w3p + (size_t)layer * 8 * 512;   // N=4 padded to 16
        v8f acc = {};
#pragma unroll
        for (int kt = 0; kt < 8; ++kt)
            acc = wmma_f16(afr[kt], load_b_frag(wL, kt, lane), acc);
        if (ncol < 4) {
            float bias = b3[layer * 4 + ncol];
#pragma unroll
            for (int v = 0; v < 8; ++v) {
                int gr = row0 + m0 + v + rhi;
                size_t idx = ((size_t)layer * BQ + gr) * 4 + ncol;
                float r  = refs[idx];
                float x  = fminf(fmaxf(r, 0.f), 1.f);
                float x1 = fmaxf(x, 1e-3f);
                float x2 = fmaxf(1.f - x, 1e-3f);
                float logit = __logf(x1 / x2);
                float val = acc[v] + bias + logit;
                out_bbox[idx] = 1.f / (1.f + __expf(-val));
            }
        }
    }
}

// ---------------------------------------------------------------------------
// Launch: repack weights into d_ws (f16 fragments, ~1.7 MB), then fused head.
// Input order: hidden_states, references, W_cls, b_cls, W1, b1, W2, b2, W3, b3
// Output: cls_scores [L,B,Q,8] then bbox_preds [L,B,Q,4], flat f32.
// ---------------------------------------------------------------------------
extern "C" void kernel_launch(void* const* d_in, const int* in_sizes, int n_in,
                              void* d_out, int out_size, void* d_ws, size_t ws_size,
                              hipStream_t stream) {
    (void)in_sizes; (void)n_in; (void)out_size; (void)ws_size;

    const float* h_in  = (const float*)d_in[0];
    const float* refs  = (const float*)d_in[1];
    const float* Wcls  = (const float*)d_in[2];
    const float* bcls  = (const float*)d_in[3];
    const float* W1    = (const float*)d_in[4];
    const float* b1    = (const float*)d_in[5];
    const float* W2    = (const float*)d_in[6];
    const float* b2    = (const float*)d_in[7];
    const float* W3    = (const float*)d_in[8];
    const float* b3    = (const float*)d_in[9];

    _Float16* ws  = (_Float16*)d_ws;
    const size_t big = (size_t)NLAYERS * 16 * 8 * 512;   // 393216 halfs
    const size_t sml = (size_t)NLAYERS * 1 * 8 * 512;    // 24576 halfs
    _Float16* w1p = ws;
    _Float16* w2p = w1p + big;
    _Float16* wcp = w2p + big;
    _Float16* w3p = wcp + sml;

    pack_weights_f16<<<NLAYERS * 16 * 8, 512, 0, stream>>>(W1,   w1p, DIM,  16, 8);
    pack_weights_f16<<<NLAYERS * 16 * 8, 512, 0, stream>>>(W2,   w2p, DIM,  16, 8);
    pack_weights_f16<<<NLAYERS *  1 * 8, 512, 0, stream>>>(Wcls, wcp, NCLS,  1, 8);
    pack_weights_f16<<<NLAYERS *  1 * 8, 512, 0, stream>>>(W3,   w3p, 4,     1, 8);

    float* out_cls  = (float*)d_out;
    float* out_bbox = out_cls + (size_t)NLAYERS * BQ * NCLS;

    spec_detr_head_kernel<<<NLAYERS * (BQ / M_TILE), BLK_THREADS, 0, stream>>>(
        h_in, refs, w1p, w2p, wcp, w3p, bcls, b1, b2, b3, out_cls, out_bbox);
}
